// GraphSAGE_72035191489045
// MI455X (gfx1250) — matrix operand order repaired and checked by reading, not compiled, not run on hardware
//
#include <hip/hip_runtime.h>

// GraphSAGE on MI455X (gfx1250): bf16 WMMA (16x16x32, f32 accum) GEMMs,
// TDM (tensor_load_to_lds) staging of the bf16 W^T tile into padded LDS,
// LDS-staged fused [self | mean(neighbors)] input tiles.

typedef __attribute__((ext_vector_type(16))) __bf16 v16bf;
typedef __attribute__((ext_vector_type(8)))  __bf16 v8bf;
typedef __attribute__((ext_vector_type(8)))  float  v8f;
typedef __attribute__((ext_vector_type(4)))  unsigned int u32x4;
typedef __attribute__((ext_vector_type(8)))  int i32x8;
typedef __attribute__((ext_vector_type(4)))  int i32x4;

#define FDIM   128           // feature dim (== H == O)
#define KDIM   256           // 2*FDIM (concat self + mean)
#define MTILE  128           // rows per workgroup
#define XS     264           // LDS row stride for X tile (KDIM + 8 pad)
#define WSTR   264           // LDS row stride for W^T tile (KDIM + 8 pad)

// One-time prep: Wt[n][k] = bf16(W[k][n]), contiguous [128][256].
__global__ __launch_bounds__(256) void wt_bf16_kernel(const float* __restrict__ W,
                                                      __bf16* __restrict__ Wt) {
    int i = blockIdx.x * 256 + threadIdx.x;    // 0 .. 32767
    int n = i >> 8;
    int k = i & 255;
    Wt[n * KDIM + k] = (__bf16)W[k * FDIM + n];
}

// out[row] = relu( concat(self_src[self_idx[row]],
//                         mean_{j<counts[row]} nbr_src[nbr_idx[row*s+j]]) @ W )
__global__ __launch_bounds__(256) void sage_layer_kernel(
    const float* __restrict__ self_src, const int* __restrict__ self_idx,
    const float* __restrict__ nbr_src,  const int* __restrict__ nbr_idx,
    const unsigned char* __restrict__ mask_self,
    const unsigned char* __restrict__ mask_nbr,
    const int* __restrict__ counts, int s,
    const __bf16* __restrict__ Wt,      // [FDIM][KDIM] bf16 (pre-transposed)
    float* __restrict__ out)            // [nrows, FDIM]
{
    extern __shared__ __bf16 smem[];
    __bf16* Xl = smem;                    // [MTILE][XS]  bf16
    __bf16* Wl = smem + MTILE * XS;       // [FDIM][WSTR] bf16 (Wl[n][k])

    const int tid     = threadIdx.x;
    const int rowbase = blockIdx.x * MTILE;

    // ---- TDM: DMA the 128x256 bf16 W^T tile global->LDS, wave 0 issues ----
    // LDS pad: insert 4 DWORDs (16B = 8 bf16) after every 128 DWORDs (512B =
    // one 256-elem row), reproducing the WSTR=264 padded layout in LDS.
    if (tid < 32) {
        unsigned long long ga = (unsigned long long)(uintptr_t)Wt;
        u32x4 g0;
        g0[0] = 1u;                                   // count=1, user descriptor
        g0[1] = (unsigned)(MTILE * XS * 2);           // lds_addr = byte offset of Wl
        g0[2] = (unsigned)(ga & 0xFFFFFFFFu);         // global_addr[31:0]
        g0[3] = (unsigned)((ga >> 32) & 0x01FFFFFFu)  // global_addr[56:32]
                | (2u << 30);                         // type = 2 ("image")
        i32x8 g1;
        g1[0] = (1 << 16)                             // data_size = 1 -> 2 bytes
              | (1 << 20)                             // pad_enable
              | (6 << 22)                             // pad_interval: 128 DWORDs
              | (3 << 25);                            // pad_amount: 4 DWORDs
        g1[1] = (int)(256u << 16);                    // tensor_dim0 = 256 (lo16)
        g1[2] = (int)(128u << 16);                    // dim0 hi = 0 | tensor_dim1 = 128 (lo16)
        g1[3] = (int)(256u << 16);                    // dim1 hi = 0 | tile_dim0 = 256
        g1[4] = 128;                                  // tile_dim1 = 128, tile_dim2 = 0
        g1[5] = 256;                                  // tensor_dim0_stride = 256 (lo32)
        g1[6] = 0;                                    // stride hi | dim1_stride lo
        g1[7] = 0;
        i32x4 gz4 = {0, 0, 0, 0};                     // groups 2/3 unused (2-D)
        i32x8 gz8 = {0, 0, 0, 0, 0, 0, 0, 0};
        __builtin_amdgcn_tensor_load_to_lds(g0, g1, gz4, gz4, gz8, 0);
    }

    // ---- stage X tile: 2 threads per row, 64 columns each ----
    {
        const int r       = tid >> 1;
        const int chalf   = tid & 1;
        const int grow    = rowbase + r;
        const int colbase = chalf * 64;

        // self half: X[r][0..127]
        long sidx = self_idx ? (long)self_idx[grow] : (long)grow;
        const float* sp = self_src + sidx * FDIM + colbase;
        const unsigned char* sm = mask_self ? (mask_self + sidx * FDIM + colbase) : nullptr;
        #pragma unroll
        for (int c = 0; c < 64; c += 4) {
            float4 v = *(const float4*)(sp + c);
            if (sm) {
                if (sm[c + 0]) v.x = -1.0f;
                if (sm[c + 1]) v.y = -1.0f;
                if (sm[c + 2]) v.z = -1.0f;
                if (sm[c + 3]) v.w = -1.0f;
            }
            Xl[r * XS + colbase + c + 0] = (__bf16)v.x;
            Xl[r * XS + colbase + c + 1] = (__bf16)v.y;
            Xl[r * XS + colbase + c + 2] = (__bf16)v.z;
            Xl[r * XS + colbase + c + 3] = (__bf16)v.w;
        }

        // neighbor-mean half: X[r][128..255]
        int cnt = counts[grow];
        if (cnt < 1) cnt = 1;
        float acc[64];
        #pragma unroll
        for (int c = 0; c < 64; ++c) acc[c] = 0.0f;

        for (int j = 0; j < cnt; ++j) {
            long nb = nbr_idx ? (long)nbr_idx[grow * s + j] : ((long)grow * s + j);
            const float* np = nbr_src + nb * FDIM + colbase;
            const unsigned char* nm = mask_nbr ? (mask_nbr + nb * FDIM + colbase) : nullptr;
            #pragma unroll
            for (int c = 0; c < 64; c += 4) {
                float4 v = *(const float4*)(np + c);
                if (nm) {
                    if (nm[c + 0]) v.x = -1.0f;
                    if (nm[c + 1]) v.y = -1.0f;
                    if (nm[c + 2]) v.z = -1.0f;
                    if (nm[c + 3]) v.w = -1.0f;
                }
                acc[c + 0] += v.x;
                acc[c + 1] += v.y;
                acc[c + 2] += v.z;
                acc[c + 3] += v.w;
            }
        }
        const float inv = 1.0f / (float)cnt;
        #pragma unroll
        for (int c = 0; c < 64; ++c)
            Xl[r * XS + FDIM + colbase + c] = (__bf16)(acc[c] * inv);
    }

    // Wave 0: make sure the TDM transfer has landed before the barrier.
    if (tid < 32) {
        __builtin_amdgcn_s_wait_tensorcnt(0);
    }
    __syncthreads();

    // ---- WMMA: wave w -> 16-row M-tile, sweep 8 N-tiles x 8 K-steps ----
    const int wave = tid >> 5;
    const int lane = tid & 31;
    const int r16  = lane & 15;
    const int half = lane >> 4;
    const int mrow = wave * 16 + r16;   // A-matrix row (same for both lane halves)

    // Preload 8 A fragments (16x32 bf16 each).
    // Layout: lanes<16 hold K {0..7, 16..23}, lanes>=16 hold K {8..15, 24..31}.
    v16bf afrag[8];
    #pragma unroll
    for (int kt = 0; kt < 8; ++kt) {
        const __bf16* base = &Xl[mrow * XS + kt * 32 + half * 8];
        v8bf lo = *(const v8bf*)(base);
        v8bf hi = *(const v8bf*)(base + 16);
        #pragma unroll
        for (int i = 0; i < 8; ++i) { afrag[kt][i] = lo[i]; afrag[kt][i + 8] = hi[i]; }
    }

    #pragma unroll
    for (int nt = 0; nt < 8; ++nt) {
        v8f c = {};
        #pragma unroll
        for (int kt = 0; kt < 8; ++kt) {
            // B fragment (32x16 bf16): lane = column, lanes<16 K {0..7, 16..23}, lanes>=16 K {8..15, 24..31}
            const __bf16* wb = &Wl[(nt * 16 + r16) * WSTR + kt * 32 + half * 8];
            v8bf b0 = *(const v8bf*)(wb);
            v8bf b1 = *(const v8bf*)(wb + 16);
            v16bf bfr;
            #pragma unroll
            for (int i = 0; i < 8; ++i) { bfr[i] = b0[i]; bfr[i + 8] = b1[i]; }
            c = __builtin_amdgcn_wmma_f32_16x16x32_bf16(
                    /*neg_a=*/false, afrag[kt], /*neg_b=*/false, bfr,
                    /*c_mod=*/(short)0, c, /*reuse_a=*/false, /*reuse_b=*/false);
        }
        // C/D layout: row = half*8 + v, col = lane&15 (within 16x16 tile)
        #pragma unroll
        for (int v = 0; v < 8; ++v) {
            float val = c[v];
            if (val < 0.0f) val = 0.0f;
            out[(long)(rowbase + wave * 16 + half * 8 + v) * FDIM + nt * 16 + r16] = val;
        }
    }
}

extern "C" void kernel_launch(void* const* d_in, const int* in_sizes, int n_in,
                              void* d_out, int out_size, void* d_ws, size_t ws_size,
                              hipStream_t stream) {
    const float*         features = (const float*)d_in[0];
    const unsigned char* mask     = (const unsigned char*)d_in[1];   // bool [V,F]
    const int*           samples0 = (const int*)d_in[2];             // [N*S0*S1]
    const int*           samples1 = (const int*)d_in[3];             // [N*S0]
    const int*           samples2 = (const int*)d_in[4];             // [N]
    const int*           counts0  = (const int*)d_in[5];             // [N*S0]
    const int*           counts1  = (const int*)d_in[6];             // [N]
    const float*         W1       = (const float*)d_in[7];           // [256,128]
    const float*         W2       = (const float*)d_in[8];           // [256,128]
    float*               out      = (float*)d_out;                   // [1024,128]

    const int N = 1024, S0 = 25, S1 = 10;
    const int rowsA = N * S0;          // 25600

    float*  a   = (float*)d_ws;                            // [25600,128] f32
    float*  b   = a + (size_t)rowsA * FDIM;                // [1024,128]  f32
    __bf16* Wt1 = (__bf16*)(b + (size_t)N * FDIM);         // [128,256]   bf16
    __bf16* Wt2 = Wt1 + (size_t)FDIM * KDIM;               // [128,256]   bf16

    const size_t ldsBytes = (size_t)(MTILE * XS + FDIM * WSTR) * sizeof(__bf16); // 132 KB

    // Pre-transpose weights to bf16 (64 KB each; stays L2-hot for the TDM DMAs).
    wt_bf16_kernel<<<(KDIM * FDIM) / 256, 256, 0, stream>>>(W1, Wt1);
    wt_bf16_kernel<<<(KDIM * FDIM) / 256, 256, 0, stream>>>(W2, Wt2);

    // a = relu(concat(mf[samples1], mean(mf[samples0], counts0)) @ W1)
    sage_layer_kernel<<<rowsA / MTILE, 256, ldsBytes, stream>>>(
        features, samples1, features, samples0, mask, mask, counts0, S1, Wt1, a);

    // b = relu(concat(mf[samples2], mean(mf[samples1], counts1)) @ W1)
    sage_layer_kernel<<<N / MTILE, 256, ldsBytes, stream>>>(
        features, samples2, features, samples1, mask, mask, counts1, S0, Wt1, b);

    // out = relu(concat(b, mean(a, counts1)) @ W2)
    sage_layer_kernel<<<N / MTILE, 256, ldsBytes, stream>>>(
        b, nullptr, a, nullptr, nullptr, nullptr, counts1, S0, Wt2, out);
}
